// Decoder_10969346474574
// MI455X (gfx1250) — matrix operand order repaired and checked by reading, not compiled, not run on hardware
//
#include <hip/hip_runtime.h>
#include <cmath>

typedef __attribute__((ext_vector_type(16))) _Float16 v16h;
typedef __attribute__((ext_vector_type(8)))  float    v8f;

#define BBATCH 2
#define NPTS   192
#define ZAT    (BBATCH*NPTS)        // 384 (z,a) rows
#define PAIRS  (ZAT*NPTS)           // 73728 pairs
#define HH     32                   // radial hidden width == WMMA K
#define NBASIS 10

#define SQ2F        1.41421356237309515f
#define SIGSCALE    1.84623f
#define Y0C         0.28209479177387814f
#define Y1C         0.4886025119029199f
#define INV_SQRT_NB 0.31622776601683794f   // 1/sqrt(10)
#define SFEAT       0.07216878364870323f   // 1/sqrt(192)

struct PathMeta { int woff, cbase, mo, mi, idim, lo; float scale; };
struct ColDesc  { unsigned char type, v, i, lf; };
struct LayerCfg {
  int ncols, ntiles, W, npaths, nscal, nvec;
  PathMeta paths[6];
  ColDesc  cols[144];
};

// ---------------- feature prescale (layer-0 input) ----------------
__global__ void k_prep(const float* __restrict__ feats, float* __restrict__ f0) {
  int i = blockIdx.x * blockDim.x + threadIdx.x;
  if (i < ZAT * 16) f0[i] = feats[i] * SFEAT;
}

// ---------------- per-pair geometry: r, Y1(3), Y2(5) --------------
__global__ void k_geom(const float* __restrict__ geo, float* __restrict__ Yb,
                       float* __restrict__ rb) {
  int p = blockIdx.x * blockDim.x + threadIdx.x;
  if (p >= PAIRS) return;
  int z = p / (NPTS * NPTS);
  int rem = p - z * NPTS * NPTS;
  int a = rem / NPTS, b = rem - a * NPTS;
  const float* ga = geo + (z * NPTS + a) * 3;
  const float* gb = geo + (z * NPTS + b) * 3;
  float dx = ga[0] - gb[0], dy = ga[1] - gb[1], dz = ga[2] - gb[2];
  float r = sqrtf(dx * dx + dy * dy + dz * dz + 1e-12f);
  float inv = 1.0f / fmaxf(r, 1e-6f);
  float x = dx * inv, y = dy * inv, zz = dz * inv;
  float* Y = Yb + p * 8;
  Y[0] = Y1C * y;  Y[1] = Y1C * zz;  Y[2] = Y1C * x;
  Y[3] = 1.0925484305920792f * x * y;
  Y[4] = 1.0925484305920792f * y * zz;
  Y[5] = 0.31539156525252005f * (2.f * zz * zz - x * x - y * y);
  Y[6] = 1.0925484305920792f * x * zz;
  Y[7] = 0.5462742152960396f * (x * x - y * y);
  rb[p] = r;
}

// ---------------- CG tensor contracted with features (per point b) ----
__global__ void k_cgfeat(LayerCfg cfg, const float* __restrict__ f0,
                         const float* __restrict__ f1, float* __restrict__ CG) {
  int zb = blockIdx.x;
  int c = threadIdx.x;
  if (c >= cfg.ncols) return;
  ColDesc d = cfg.cols[c];
  const float INV3 = 0.5773502691896258f, INV6 = 0.4082482904638631f;
  const float ALP = 0.31622776601683794f, BET = 0.18257418583505536f;
  float m0 = 0.f, m1 = 0.f, m2 = 0.f, m3 = 0.f, m4 = 0.f;
  if (d.type == 0) {                       // (0,0,0): c000 = 1
    m0 = f0[zb * 16 + d.v];
  } else if (d.type == 1) {                // (1,0,1): delta(i,m)/sqrt3
    float fv = f0[zb * 16 + d.v] * INV3;
    if (d.i == 0) m0 = fv; else if (d.i == 1) m1 = fv; else m2 = fv;
  } else {
    const float* f = f1 + (zb * 8 + d.v) * 3;
    float fy = f[0], fz = f[1], fx = f[2]; // l=1 order (y,z,x)
    if (d.type == 2) {                     // (0,1,1): delta(j,m)/sqrt3
      m0 = fy * INV3; m1 = fz * INV3; m2 = fx * INV3;
    } else if (d.type == 3) {              // (1,1,0): delta(i,j)/sqrt3
      m0 = f[d.i] * INV3;
    } else if (d.type == 4) {              // (1,1,1): eps/sqrt6
      if (d.i == 0)      { m1 = -fx * INV6; m2 =  fz * INV6; }
      else if (d.i == 1) { m0 =  fx * INV6; m2 = -fy * INV6; }
      else               { m0 = -fz * INV6; m1 =  fy * INV6; }
    } else {                               // (1,1,2): c112 normalized
      if (d.i == 0)      { m0 = ALP * fx; m1 = ALP * fz; m2 = -BET * fy; m4 = -ALP * fy; }
      else if (d.i == 1) { m1 = ALP * fy; m2 = 2.f * BET * fz; m3 = ALP * fx; }
      else               { m0 = ALP * fy; m2 = -BET * fx; m3 = ALP * fz; m4 = ALP * fx; }
    }
  }
  float* o = CG + (zb * cfg.ncols + c) * 5;
  o[0] = m0; o[1] = m1; o[2] = m2; o[3] = m3; o[4] = m4;
}

// ---------------- radial MLP -> A fragments (16-bit A layout) ----------
__global__ __launch_bounds__(256) void k_hmlp(const float* __restrict__ rb,
                                              const float* __restrict__ w0,
                                              const float* __restrict__ b0,
                                              _Float16* __restrict__ Afrag) {
  __shared__ float sw0[NBASIS * HH];
  __shared__ float sb0[HH];
  for (int i = threadIdx.x; i < NBASIS * HH; i += blockDim.x) sw0[i] = w0[i];
  if (threadIdx.x < HH) sb0[threadIdx.x] = b0[threadIdx.x];
  __syncthreads();
  int p = blockIdx.x * blockDim.x + threadIdx.x;
  if (p >= PAIRS) return;
  float r = rb[p];
  float basis[NBASIS];
#pragma unroll
  for (int k = 0; k < NBASIS; k++) {
    float dg = r * 2.25f - (float)k;                 // (r - k*step)/step, step=4/9
    float cv = __cosf(1.5707963267948966f * dg);
    basis[k] = (fabsf(dg) < 1.0f) ? cv * cv : 0.0f;
  }
  float acc[HH];
#pragma unroll
  for (int j = 0; j < HH; j++) acc[j] = 0.f;
#pragma unroll
  for (int k = 0; k < NBASIS; k++) {
    float bk = basis[k];
#pragma unroll
    for (int j = 0; j < HH; j++) acc[j] += bk * sw0[k * HH + j];
  }
  int za = p / NPTS, b = p - za * NPTS;
  int t = b >> 5, kk = b & 31;
  int laneHalf = (kk >> 3) & 1;                      // K 8-15,24-31 -> lanes 16-31
  int halfpos = ((kk & 16) >> 1) | (kk & 7);         // dword*2+half within lane
#pragma unroll
  for (int j = 0; j < HH; j++) {
    float hv = fmaxf(acc[j] * INV_SQRT_NB + sb0[j], 0.0f) * SQ2F;
    int mtile = j >> 4;
    int lane = (j & 15) + (laneHalf << 4);
    Afrag[((((za * 2 + mtile) * 6 + t) * 32 + lane) << 4) + halfpos] = (_Float16)hv;
  }
}

// ---------------- Q = Y . CGfeat -> B fragments ------------------------
__global__ __launch_bounds__(256) void k_qfrag(LayerCfg cfg, const float* __restrict__ Yb,
                                               const float* __restrict__ CG,
                                               _Float16* __restrict__ Bfrag) {
  int zat = blockIdx.x;                 // za*6 + t
  int za = zat / 6, t = zat - za * 6;
  int z = za / NPTS;
  int total = cfg.ncols * 32;
  for (int idx = threadIdx.x; idx < total; idx += blockDim.x) {
    int c = idx >> 5, kk = idx & 31;
    int b = t * 32 + kk;
    const float* cg = CG + ((z * NPTS + b) * cfg.ncols + c) * 5;
    const float* Y = Yb + (za * NPTS + b) * 8;
    int lf = cfg.cols[c].lf;
    float q;
    if (lf == 0)      q = Y0C * cg[0];
    else if (lf == 1) q = Y[0] * cg[0] + Y[1] * cg[1] + Y[2] * cg[2];
    else              q = Y[3] * cg[0] + Y[4] * cg[1] + Y[5] * cg[2] + Y[6] * cg[3] + Y[7] * cg[4];
    int lane = (c & 15) + (((kk >> 3) & 1) << 4);    // N col in lane, K 8-15,24-31 -> hi half
    Bfrag[(((zat * cfg.ntiles + (c >> 4)) * 32 + lane) << 4) + (((kk & 16) >> 1) | (kk & 7))] = (_Float16)q;
  }
}

// ---------------- S[h,c] = sum_b h * Q : WMMA GEMM, K=192 --------------
// One wave owns MT m-tiles x NT n-tiles of a za row: MT*NT >= 8 independent
// accumulators keep dependent WMMAs outside the 5-slot f16 RAW hazard window
// (no v_nop padding); per k-step all fragments are batch-loaded first so the
// load clause overlaps the WMMA burst; A/B fragments are register-reused
// across the tile row/column.
template <int NT, int MT>
__global__ __launch_bounds__(128) void k_wmma(const _Float16* __restrict__ Afrag,
                                              const _Float16* __restrict__ Bfrag,
                                              float* __restrict__ Sfrag, int njobs) {
  int lane = threadIdx.x & 31;
  int job = blockIdx.x * (blockDim.x >> 5) + (threadIdx.x >> 5);
  if (job >= njobs) return;               // wave-uniform guard (EXEC all-ones)
  int za, mbase;
  if (MT == 2) { za = job;      mbase = 0; }
  else         { za = job >> 1; mbase = job & 1; }
  v8f acc[MT][NT];
#pragma unroll
  for (int m = 0; m < MT; m++)
#pragma unroll
    for (int n = 0; n < NT; n++)
      acc[m][n] = (v8f){0.f, 0.f, 0.f, 0.f, 0.f, 0.f, 0.f, 0.f};
#pragma unroll
  for (int t = 0; t < 6; t++) {
    const _Float16* bp = Bfrag + (((size_t)(za * 6 + t) * NT * 32 + lane) << 4);
    v16h av[MT];
#pragma unroll
    for (int m = 0; m < MT; m++)
      av[m] = *(const v16h*)(Afrag +
                             ((((za * 2 + mbase + m) * 6 + t) * 32 + lane) << 4));
    v16h bvs[NT];
#pragma unroll
    for (int n = 0; n < NT; n++)
      bvs[n] = *(const v16h*)(bp + ((size_t)n * 32 << 4));
    if (t < 5) {                          // pull next k-step toward the WGP
#pragma unroll
      for (int m = 0; m < MT; m++)
        __builtin_prefetch(Afrag +
                           ((((za * 2 + mbase + m) * 6 + t + 1) * 32 + lane) << 4), 0, 3);
      __builtin_prefetch(bp + ((size_t)NT * 32 << 4), 0, 3);
    }
#pragma unroll
    for (int n = 0; n < NT; n++)
#pragma unroll
      for (int m = 0; m < MT; m++)
        acc[m][n] = __builtin_amdgcn_wmma_f32_16x16x32_f16(false, av[m], false, bvs[n],
                                                           (short)0, acc[m][n],
                                                           false, false);
  }
#pragma unroll
  for (int m = 0; m < MT; m++)
#pragma unroll
    for (int n = 0; n < NT; n++)
      *(v8f*)(Sfrag + ((((za * 2 + mbase + m) * NT + n) * 32 + lane) << 3)) = acc[m][n];
}

// ---------------- wout contraction + gating / final output -------------
__global__ __launch_bounds__(64) void k_out(LayerCfg cfg, const float* __restrict__ Sfrag,
                                            const float* __restrict__ wout,
                                            float* __restrict__ f0n, float* __restrict__ f1n,
                                            float* __restrict__ outp, int isFinal, int m0real) {
  __shared__ float cvals[64];
  int za = blockIdx.x;
  int nout = cfg.nscal + 3 * cfg.nvec;
  int o = threadIdx.x;
  if (o < nout) {
    int lo, u, i;
    if (o < cfg.nscal) { lo = 0; u = o; i = 0; }
    else { int oo = o - cfg.nscal; lo = 1; u = oo / 3; i = oo - u * 3; }
    float acc = 0.f;
    for (int p = 0; p < cfg.npaths; p++) {
      PathMeta pm = cfg.paths[p];
      if (pm.lo != lo) continue;
      float pa = 0.f;
      for (int h = 0; h < HH; h++) {
        const float* wr = wout + h * cfg.W + pm.woff + u * pm.mi;
        int hr = h & 15, mt = h >> 4, rv = hr & 7, lhb = (hr >> 3) << 4;
        for (int v = 0; v < pm.mi; v++) {
          int c = pm.cbase + v * pm.idim + i;
          int lane = lhb | (c & 15);
          int ntile = c >> 4;
          pa += wr[v] *
                Sfrag[((((za * 2 + mt) * cfg.ntiles + ntile) * 32 + lane) << 3) + rv];
        }
      }
      acc += pa * pm.scale;
    }
    cvals[o] = acc;
  }
  __syncthreads();
  if (o >= nout) return;
  if (isFinal) {
    if (o < cfg.nscal) outp[za * 8 + o * 4] = cvals[o];
    else {
      int oo = o - cfg.nscal; int u = oo / 3, i = oo - u * 3;
      outp[za * 8 + u * 4 + 1 + i] = cvals[o];
    }
  } else {
    if (o < m0real) {
      f0n[za * 16 + o] = fmaxf(cvals[o], 0.f) * SQ2F * SFEAT;
    } else if (o >= cfg.nscal) {
      int oo = o - cfg.nscal; int u = oo / 3, i = oo - u * 3;
      float g = SIGSCALE / (1.f + __expf(-cvals[m0real + u]));
      f1n[(za * 8 + u) * 3 + i] = cvals[o] * g * SFEAT;
    }
  }
}

// ---------------- host: layer descriptors ------------------------------
static void build_cfg(int layer, LayerCfg& cfg) {
  struct P { int lo, li, lf, mo, mi; };
  P pl[6]; int np;
  if (layer == 0) {
    np = 2;
    pl[0] = {0, 0, 0, 24, 16}; pl[1] = {1, 0, 1, 8, 16};
  } else {
    int mo0 = (layer == 1) ? 24 : 2, mo1 = (layer == 1) ? 8 : 2;
    np = 6;
    pl[0] = {0, 0, 0, mo0, 16}; pl[1] = {0, 1, 1, mo0, 8};
    pl[2] = {1, 0, 1, mo1, 16}; pl[3] = {1, 1, 0, mo1, 8};
    pl[4] = {1, 1, 1, mo1, 8};  pl[5] = {1, 1, 2, mo1, 8};
  }
  int npl[2] = {0, 0};
  for (int p = 0; p < np; p++) npl[pl[p].lo]++;
  const double invsqrtH = 1.0 / sqrt(32.0);
  int woff = 0, cbase = 0;
  cfg.npaths = np;
  for (int p = 0; p < np; p++) {
    int idim = 2 * pl[p].lo + 1;
    PathMeta& m = cfg.paths[p];
    m.woff = woff; m.cbase = cbase; m.mo = pl[p].mo; m.mi = pl[p].mi;
    m.idim = idim; m.lo = pl[p].lo;
    m.scale = (float)(invsqrtH / sqrt((double)(pl[p].mi * npl[pl[p].lo])));
    int type;
    if (pl[p].li == 0) type = (pl[p].lf == 0) ? 0 : 1;
    else if (pl[p].lo == 0) type = 2;
    else type = 3 + pl[p].lf;
    for (int v = 0; v < pl[p].mi; v++)
      for (int i = 0; i < idim; i++) {
        ColDesc& d = cfg.cols[cbase + v * idim + i];
        d.type = (unsigned char)type; d.v = (unsigned char)v;
        d.i = (unsigned char)i; d.lf = (unsigned char)pl[p].lf;
      }
    woff += pl[p].mo * pl[p].mi;
    cbase += pl[p].mi * idim;
  }
  cfg.W = woff; cfg.ncols = cbase; cfg.ntiles = cbase / 16;
  cfg.nscal = (layer == 2) ? 2 : 24;
  cfg.nvec = (layer == 2) ? 2 : 8;
}

extern "C" void kernel_launch(void* const* d_in, const int* in_sizes, int n_in,
                              void* d_out, int out_size, void* d_ws, size_t ws_size,
                              hipStream_t stream) {
  (void)in_sizes; (void)n_in; (void)out_size; (void)ws_size;
  const float* feats = (const float*)d_in[0];
  const float* geo = (const float*)d_in[1];
  const float* w0s[3]   = {(const float*)d_in[2], (const float*)d_in[5], (const float*)d_in[8]};
  const float* b0s[3]   = {(const float*)d_in[3], (const float*)d_in[6], (const float*)d_in[9]};
  const float* wouts[3] = {(const float*)d_in[4], (const float*)d_in[7], (const float*)d_in[10]};

  char* ws = (char*)d_ws;
  size_t off = 0;
  auto alloc = [&](size_t bytes) -> char* {
    char* p = ws + off;
    off = (off + bytes + 255) & ~(size_t)255;
    return p;
  };
  float*    CG = (float*)alloc((size_t)ZAT * 144 * 5 * 4);
  float*    Yb = (float*)alloc((size_t)PAIRS * 8 * 4);
  float*    rb = (float*)alloc((size_t)PAIRS * 4);
  _Float16* Af = (_Float16*)alloc((size_t)ZAT * 2 * 6 * 32 * 16 * 2);
  _Float16* Bf = (_Float16*)alloc((size_t)ZAT * 6 * 9 * 32 * 16 * 2);
  float*    Sf = (float*)alloc((size_t)ZAT * 2 * 9 * 32 * 8 * 4);
  float*    f0a = (float*)alloc((size_t)ZAT * 16 * 4);
  float*    f1a = (float*)alloc((size_t)ZAT * 8 * 3 * 4);
  float*    f0b = (float*)alloc((size_t)ZAT * 16 * 4);
  float*    f1b = (float*)alloc((size_t)ZAT * 8 * 3 * 4);

  k_prep<<<(ZAT * 16 + 255) / 256, 256, 0, stream>>>(feats, f0a);
  k_geom<<<(PAIRS + 255) / 256, 256, 0, stream>>>(geo, Yb, rb);

  float* f0c = f0a; float* f1c = f1a; float* f0n = f0b; float* f1n = f1b;
  for (int layer = 0; layer < 3; layer++) {
    LayerCfg cfg;
    build_cfg(layer, cfg);
    k_cgfeat<<<ZAT, 160, 0, stream>>>(cfg, f0c, f1c, CG);
    k_hmlp<<<(PAIRS + 255) / 256, 256, 0, stream>>>(rb, w0s[layer], b0s[layer], Af);
    k_qfrag<<<ZAT * 6, 256, 0, stream>>>(cfg, Yb, CG, Bf);
    if (cfg.ntiles == 4) {
      int njobs = ZAT;                    // one wave per za row (both mtiles)
      k_wmma<4, 2><<<(njobs + 3) / 4, 128, 0, stream>>>(Af, Bf, Sf, njobs);
    } else {
      int njobs = ZAT * 2;                // one wave per (za, mtile) row
      k_wmma<9, 1><<<(njobs + 3) / 4, 128, 0, stream>>>(Af, Bf, Sf, njobs);
    }
    k_out<<<ZAT, 64, 0, stream>>>(cfg, Sf, wouts[layer], f0n, f1n, (float*)d_out,
                                  layer == 2 ? 1 : 0, 16);
    float* t0 = f0c; f0c = f0n; f0n = t0;
    float* t1 = f1c; f1c = f1n; f1n = t1;
  }
}
